// LeadGNNModel_87711822118991
// MI455X (gfx1250) — matrix operand order repaired and checked by reading, not compiled, not run on hardware
//
#include <hip/hip_runtime.h>
#include <hip/hip_bf16.h>

// ---------------------------------------------------------------------------
// GCN model for MI455X (gfx1250, wave32, WMMA).
//
// Shapes: B=128, N=2048, E=16384, F=32, H=64, OUT=16.
// Dense GEMMs use v_wmma_f32_16x16x32_f16 (f16 in, f32 accumulate).
// Scatter-adds use global_atomic_add_f32; the 67MB agg buffer fits in the
// 192MB L2, so atomics stay on-chip. Total HBM traffic ~300MB -> bandwidth
// bound at 23.3 TB/s; GEMM FLOPs (3.2G) are negligible for the WMMA pipes.
//
// Workspace layout (bytes):
//   [0,   1KB)            : int n[B]        (valid node count per batch)
//   [1KB, +1MB)           : float deg[B*N]
//   next  1MB             : float dis[B*N]
//   next  67MB            : float bufA[B*N*64]  (hW of current layer)
//   next  67MB            : float bufB[B*N*64]  (agg / layer output)
// Total ~136.3 MB.
// ---------------------------------------------------------------------------

#define Bc   128
#define Nc   2048
#define Ec   16384
#define Fc   32
#define Hc   64
#define OUTc 16
#define BNc  (Bc * Nc)

typedef __attribute__((ext_vector_type(16))) _Float16 v16h;
typedef __attribute__((ext_vector_type(8)))  _Float16 v8h;
typedef __attribute__((ext_vector_type(8)))  float    v8f;

// ---------------------------------------------------------------------------
// n[b] = sum(mask[b,:])   (mask is monotone true-prefix, but we just sum)
// ---------------------------------------------------------------------------
__global__ __launch_bounds__(256)
void count_kernel(const unsigned char* __restrict__ mask, int* __restrict__ nB)
{
    __shared__ int red[256];
    const int b = blockIdx.x, t = threadIdx.x;
    int s = 0;
    for (int i = t; i < Nc; i += 256) s += mask[b * Nc + i] ? 1 : 0;
    red[t] = s;
    __syncthreads();
    for (int off = 128; off > 0; off >>= 1) {
        if (t < off) red[t] += red[t + off];
        __syncthreads();
    }
    if (t == 0) nB[b] = red[0];
}

// deg init: deg = mask (the "+ node_mask" self-loop term of the reference)
__global__ __launch_bounds__(256)
void deg_init_kernel(const unsigned char* __restrict__ mask, float* __restrict__ deg)
{
    const int i = blockIdx.x * 256 + threadIdx.x;
    if (i < BNc) deg[i] = mask[i] ? 1.f : 0.f;
}

// deg += segment_sum(edge weights)
__global__ __launch_bounds__(256)
void edge_deg_kernel(const int* __restrict__ ei, const int* __restrict__ nB,
                     float* __restrict__ deg)
{
    const int idx = blockIdx.x * 256 + threadIdx.x;
    if (idx >= Bc * Ec) return;
    const int b = idx >> 14;          // /E
    const int e = idx & (Ec - 1);
    const int* base = ei + b * 2 * Ec;
    const int s = base[e];
    const int d = base[Ec + e];
    const int mx = nB[b] - 1;
    if (s <= mx && d <= mx)
        atomicAdd(&deg[b * Nc + d], 1.f);
}

__global__ __launch_bounds__(256)
void dis_kernel(const float* __restrict__ deg, float* __restrict__ dis)
{
    const int i = blockIdx.x * 256 + threadIdx.x;
    if (i < BNc) {
        const float d = deg[i];
        dis[i] = d > 0.f ? rsqrtf(d) : 0.f;
    }
}

// ---------------------------------------------------------------------------
// hW = (X * mask) @ W     using v_wmma_f32_16x16x32_f16.
//   X: (B*N) x K row-major f32,  W: K x 64 row-major f32, out: (B*N) x 64 f32.
// Block = 256 threads = 8 wave32s; block covers 128 rows; wave owns 16 rows.
// LDS: sX = masked X in f16 (row-major), sW = W transposed in f16 (sW[n*K+k]).
//
// Per-lane fragment construction per ISA 7.12.2 (wave32):
//   A (16x32 f16): lane l<16 row M=l holds K = {hi*8+e} for e<8 and
//                  {16+hi*8+e} for e>=8, hi = lane>>4  -> two 16B LDS runs.
//   B (32x16 f16): same K order as A, N=lane&15
//                  -> two 16B runs of W^T.
//   C/D (16x16 f32): VGPR v holds row (hi*8 + v), col lane&15.
// ---------------------------------------------------------------------------
template <int K>
__global__ __launch_bounds__(256)
void gemm_mask_kernel(const float* __restrict__ X, const float* __restrict__ W,
                      const unsigned char* __restrict__ mask,
                      float* __restrict__ out)
{
    __shared__ __align__(16) _Float16 sX[128 * K];
    __shared__ __align__(16) _Float16 sW[Hc * K];

    const int t = threadIdx.x;
    const int rowBlock = blockIdx.x * 128;

    // Stage W^T into LDS (f32 -> f16). Coalesced global reads.
    for (int i = t; i < Hc * K; i += 256) {
        const int k = i >> 6;        // / 64
        const int n = i & 63;
        sW[n * K + k] = (_Float16)W[i];
    }
    // Stage masked X into LDS. Consecutive t -> consecutive k: coalesced b128s.
    for (int i = t; i < 128 * K; i += 256) {
        const int r = i / K;
        const int k = i % K;
        const int row = rowBlock + r;
        const float m = mask[row] ? 1.f : 0.f;
        sX[r * K + k] = (_Float16)(X[row * K + k] * m);
    }
    __syncthreads();

    const int wave = t >> 5, lane = t & 31;
    const int l = lane & 15, hi = lane >> 4;
    const int rowbase = wave * 16;

    v8f acc[4];
#pragma unroll
    for (int nt = 0; nt < 4; ++nt)
#pragma unroll
        for (int v = 0; v < 8; ++v) acc[nt][v] = 0.f;

#pragma unroll
    for (int c = 0; c < K / 32; ++c) {
        // A fragment: two 8-half (16B) LDS loads.
        v16h a;
        {
            const v8h lo = *(const v8h*)&sX[(rowbase + l) * K + c * 32 + hi * 8];
            const v8h hp = *(const v8h*)&sX[(rowbase + l) * K + c * 32 + 16 + hi * 8];
#pragma unroll
            for (int e = 0; e < 8; ++e) { a[e] = lo[e]; a[8 + e] = hp[e]; }
        }
#pragma unroll
        for (int nt = 0; nt < 4; ++nt) {
            const int n = nt * 16 + l;
            const v8h* q = (const v8h*)&sW[n * K + c * 32 + hi * 8];
            const v8h b0 = q[0], b1v = q[2];
            v16h bfrag;
#pragma unroll
            for (int e = 0; e < 8; ++e) { bfrag[e] = b0[e]; bfrag[8 + e] = b1v[e]; }
            acc[nt] = __builtin_amdgcn_wmma_f32_16x16x32_f16(
                false, a, false, bfrag, (short)0, acc[nt], false, false);
        }
    }

    // Write D: row = rowBlock + rowbase + hi*8 + v, col = nt*16 + l.
#pragma unroll
    for (int nt = 0; nt < 4; ++nt) {
#pragma unroll
        for (int v = 0; v < 8; ++v) {
            const int row = rowBlock + rowbase + hi * 8 + v;
            out[row * Hc + nt * 16 + l] = acc[nt][v];
        }
    }
}

// agg = hW * dis^2 * mask     (self-loop term)
__global__ __launch_bounds__(256)
void agg_init_kernel(const float* __restrict__ hW, const float* __restrict__ dis,
                     const unsigned char* __restrict__ mask, float* __restrict__ agg)
{
    const int idx = blockIdx.x * 256 + threadIdx.x;
    if (idx >= BNc * Hc) return;
    const int node = idx >> 6;
    const float dd = dis[node];
    const float m = mask[node] ? 1.f : 0.f;
    agg[idx] = hW[idx] * dd * dd * m;
}

// agg[dst,c] += hW[src,c] * dis[src]*dis[dst]   for valid edges.
// One thread per (edge, channel); 64 consecutive threads share an edge so the
// atomics are fully coalesced and land in L2 (agg fits in 192MB L2).
__global__ __launch_bounds__(256)
void scatter_kernel(const int* __restrict__ ei, const int* __restrict__ nB,
                    const float* __restrict__ dis, const float* __restrict__ hW,
                    float* __restrict__ agg)
{
    const long idx = (long)blockIdx.x * 256 + threadIdx.x;
    if (idx >= (long)Bc * Ec * Hc) return;
    const int c  = (int)(idx & 63);
    const int eg = (int)(idx >> 6);           // global edge index
    const int b  = eg >> 14;                  // / E
    const int e  = eg & (Ec - 1);
    const int* base = ei + b * 2 * Ec;
    const int s = base[e];
    const int d = base[Ec + e];
    const int mx = nB[b] - 1;
    if (s > mx || d > mx) return;
    const float norm = dis[b * Nc + s] * dis[b * Nc + d];
    atomicAdd(&agg[(b * Nc + d) * Hc + c], hW[(b * Nc + s) * Hc + c] * norm);
}

// h = relu((agg + bias) * mask)   (in place over agg buffer)
__global__ __launch_bounds__(256)
void finalize_kernel(float* __restrict__ agg, const float* __restrict__ bias,
                     const unsigned char* __restrict__ mask)
{
    const int idx = blockIdx.x * 256 + threadIdx.x;
    if (idx >= BNc * Hc) return;
    const int node = idx >> 6;
    const int c = idx & 63;
    const float m = mask[node] ? 1.f : 0.f;
    agg[idx] = fmaxf((agg[idx] + bias[c]) * m, 0.f);
}

// Mean-pool over nodes + tiny 64->64->16 MLP (one block per batch element).
__global__ __launch_bounds__(256)
void pool_mlp_kernel(const float* __restrict__ h2, const int* __restrict__ nB,
                     const float* __restrict__ aW1, const float* __restrict__ ab1,
                     const float* __restrict__ aW2, const float* __restrict__ ab2,
                     float* __restrict__ outp)
{
    __shared__ float red[256];
    __shared__ float pooled[Hc];
    __shared__ float hidden[Hc];
    const int b = blockIdx.x, t = threadIdx.x;
    const int c = t & 63, seg = t >> 6;
    float s = 0.f;
    for (int n = seg; n < Nc; n += 4)
        s += h2[(b * Nc + n) * Hc + c];     // coalesced across t
    red[t] = s;
    __syncthreads();
    if (t < Hc) {
        const int nb = nB[b];
        const float inv = 1.f / (float)(nb > 1 ? nb : 1);
        pooled[t] = (red[t] + red[t + 64] + red[t + 128] + red[t + 192]) * inv;
    }
    __syncthreads();
    if (t < Hc) {
        float acc = ab1[t];
#pragma unroll 8
        for (int k = 0; k < Hc; ++k) acc += pooled[k] * aW1[k * Hc + t];
        hidden[t] = fmaxf(acc, 0.f);
    }
    __syncthreads();
    if (t < OUTc) {
        float acc = ab2[t];
#pragma unroll 8
        for (int k = 0; k < Hc; ++k) acc += hidden[k] * aW2[k * OUTc + t];
        outp[b * OUTc + t] = acc;
    }
}

// ---------------------------------------------------------------------------
extern "C" void kernel_launch(void* const* d_in, const int* in_sizes, int n_in,
                              void* d_out, int out_size, void* d_ws, size_t ws_size,
                              hipStream_t stream)
{
    const float*         x    = (const float*)d_in[0];
    const int*           ei   = (const int*)d_in[1];
    const unsigned char* mask = (const unsigned char*)d_in[2];   // jnp bool -> 1 byte
    const float*         W1   = (const float*)d_in[3];
    const float*         b1   = (const float*)d_in[4];
    const float*         W2   = (const float*)d_in[5];
    const float*         b2   = (const float*)d_in[6];
    const float*         aW1  = (const float*)d_in[7];
    const float*         ab1  = (const float*)d_in[8];
    const float*         aW2  = (const float*)d_in[9];
    const float*         ab2  = (const float*)d_in[10];
    float*               out  = (float*)d_out;

    char*  ws   = (char*)d_ws;
    int*   nB   = (int*)ws;
    float* deg  = (float*)(ws + 1024);
    float* dis  = deg + BNc;
    float* bufA = dis + BNc;            // hW
    float* bufB = bufA + (long)BNc * Hc; // agg / layer output

    const int elemGrid = (BNc * Hc) / 256;      // 65536
    const int edgeGrid = (Bc * Ec) / 256;       // 8192
    const int scatGrid = (int)(((long)Bc * Ec * Hc) / 256); // 524288

    // normalization
    count_kernel<<<Bc, 256, 0, stream>>>(mask, nB);
    deg_init_kernel<<<BNc / 256, 256, 0, stream>>>(mask, deg);
    edge_deg_kernel<<<edgeGrid, 256, 0, stream>>>(ei, nB, deg);
    dis_kernel<<<BNc / 256, 256, 0, stream>>>(deg, dis);

    // layer 1: K = F = 32
    gemm_mask_kernel<Fc><<<BNc / 128, 256, 0, stream>>>(x, W1, mask, bufA);
    agg_init_kernel<<<elemGrid, 256, 0, stream>>>(bufA, dis, mask, bufB);
    scatter_kernel<<<scatGrid, 256, 0, stream>>>(ei, nB, dis, bufA, bufB);
    finalize_kernel<<<elemGrid, 256, 0, stream>>>(bufB, b1, mask);

    // layer 2: K = H = 64
    gemm_mask_kernel<Hc><<<BNc / 128, 256, 0, stream>>>(bufB, W2, mask, bufA);
    agg_init_kernel<<<elemGrid, 256, 0, stream>>>(bufA, dis, mask, bufB);
    scatter_kernel<<<scatGrid, 256, 0, stream>>>(ei, nB, dis, bufA, bufB);
    finalize_kernel<<<elemGrid, 256, 0, stream>>>(bufB, b2, mask);

    // pooling + MLP head
    pool_mlp_kernel<<<Bc, 256, 0, stream>>>(bufB, nB, aW1, ab1, aW2, ab2, out);
}
